// EnergyGatedDeltaModel_3204045603493
// MI455X (gfx1250) — matrix operand order repaired and checked by reading, not compiled, not run on hardware
//
#include <hip/hip_runtime.h>
#include <hip/hip_bf16.h>

typedef __attribute__((ext_vector_type(16))) _Float16 v16h;
typedef __attribute__((ext_vector_type(8)))  float    v8f;

#define BATCH  512
#define SEQLEN 2048
#define HID    32
#define VOC    64

union AFrag { v16h v; _Float16 h[16]; unsigned u[8]; };
union CFrag { v8f  v; float f[8]; };

// may-alias 4-byte LDS pair load (two consecutive f16 -> one dword)
__device__ __forceinline__ unsigned lds_pair(const _Float16* p) {
    unsigned r; __builtin_memcpy(&r, p, 4); return r;
}

// lane broadcast via v_readlane_b32 (constant lane -> SGPR result, no LDS pipe)
__device__ __forceinline__ float bcastf(float v, int l) {
    return __int_as_float(__builtin_amdgcn_readlane(__float_as_int(v), l));
}

// ---------------------------------------------------------------------------
// Kernel 1: encoder.  16 tokens per WMMA tile, 1 wave per tile (grid-stride).
// GEMM1: (16x32 f16) x (32x64) via 4x v_wmma_f32_16x16x32_f16
// GEMM2: (16x64 f16) x (64x32) via 4x v_wmma (2 K-chunks x 2 N-tiles)
// ---------------------------------------------------------------------------
__global__ __launch_bounds__(256) void encoder_kernel(
    const int*   __restrict__ seq,  const float* __restrict__ embed,
    const float* __restrict__ w1,   const float* __restrict__ b1,
    const float* __restrict__ w2,   const float* __restrict__ b2,
    const float* __restrict__ lng,  const float* __restrict__ lnb,
    float* __restrict__ hidden)
{
    __shared__ _Float16 embedH[VOC * HID];   // [v][h] row-major, f16
    __shared__ _Float16 w1T[VOC * HID];      // transposed: [n(64)][k(32)]
    __shared__ _Float16 w2T[HID * VOC];      // transposed: [n(32)][k(64)]
    __shared__ float b1f[2 * HID], b2f[HID], lngf[HID], lnbf[HID];
    struct WaveScr {
        union { _Float16 ffl[16 * 64]; float outb[16 * 32]; } u; // 2 KB, reused
        float xbuf[16 * 32];                                     // 2 KB
        int   idx[16];
    };
    __shared__ WaveScr wsh[8];

    const int tid = threadIdx.x;
    for (int i = tid; i < VOC * HID; i += 256) embedH[i] = (_Float16)embed[i];
    for (int i = tid; i < 2048; i += 256) { int n = i >> 5, k = i & 31; w1T[i] = (_Float16)w1[k * 64 + n]; }
    for (int i = tid; i < 2048; i += 256) { int n = i >> 6, k = i & 63; w2T[i] = (_Float16)w2[k * 32 + n]; }
    if (tid < 64) b1f[tid] = b1[tid];
    if (tid < 32) { b2f[tid] = b2[tid]; lngf[tid] = lng[tid]; lnbf[tid] = lnb[tid]; }
    __syncthreads();

    const int wid = tid >> 5, lane = tid & 31;
    const int m16 = lane & 15, hi = lane >> 4;   // ISA 16-bit fragment layout helpers
    WaveScr& S = wsh[wid];

    // B fragments (live for the whole kernel): element e -> k = e + 16*hi, n = lane%16
    AFrag w1f[4], w2f[2][2];
    #pragma unroll
    for (int t = 0; t < 4; ++t) {
        int n = t * 16 + m16;
        #pragma unroll
        for (int p = 0; p < 8; ++p) w1f[t].u[p] = lds_pair(&w1T[n * 32 + (2 * p + 16 * hi)]);
    }
    #pragma unroll
    for (int s = 0; s < 2; ++s)
        #pragma unroll
        for (int t = 0; t < 2; ++t) {
            int n = t * 16 + m16;
            #pragma unroll
            for (int p = 0; p < 8; ++p)
                w2f[s][t].u[p] = lds_pair(&w2T[n * 64 + (2 * p + 16 * hi + 32 * s)]);
        }

    const int nTiles = (BATCH * SEQLEN) / 16;
    const int nWaves = gridDim.x * 8;
    for (int tile = blockIdx.x * 8 + wid; tile < nTiles; tile += nWaves) {
        int idx = seq[tile * 16 + m16];
        if (lane < 16) S.idx[lane] = idx;

        // A1 fragment: row m = lane%16, K = 0..15 + 16*hi
        AFrag a1;
        #pragma unroll
        for (int p = 0; p < 8; ++p) {
            int k = 2 * p + 16 * hi;
            a1.u[p] = lds_pair(&embedH[idx * HID + k]);
        }

        // GEMM1: ff1 = h @ w1
        CFrag c1[4];
        #pragma unroll
        for (int t = 0; t < 4; ++t) {
            v8f z = {};
            c1[t].v = __builtin_amdgcn_wmma_f32_16x16x32_f16(false, a1.v, false, w1f[t].v,
                                                             (short)0, z, false, false);
        }
        // bias + relu -> f16 staging (row-major [m][64])
        #pragma unroll
        for (int t = 0; t < 4; ++t) {
            int n = t * 16 + m16;
            float bb = b1f[n];
            #pragma unroll
            for (int r = 0; r < 8; ++r) {
                int m = r + 8 * hi;
                float v = c1[t].f[r] + bb;
                S.u.ffl[m * 64 + n] = (_Float16)(v > 0.f ? v : 0.f);
            }
        }
        // A2 fragments (K = 0..31 and 32..63)
        AFrag a2[2];
        #pragma unroll
        for (int s = 0; s < 2; ++s)
            #pragma unroll
            for (int p = 0; p < 8; ++p) {
                int k = 2 * p + 16 * hi + 32 * s;
                a2[s].u[p] = lds_pair(&S.u.ffl[m16 * 64 + k]);
            }
        // GEMM2: ff2 = relu(ff1) @ w2   (chained accumulation over K-chunks)
        CFrag c2[2];
        #pragma unroll
        for (int t = 0; t < 2; ++t) {
            v8f z = {};
            v8f acc = __builtin_amdgcn_wmma_f32_16x16x32_f16(false, a2[0].v, false, w2f[0][t].v,
                                                             (short)0, z, false, false);
            c2[t].v  = __builtin_amdgcn_wmma_f32_16x16x32_f16(false, a2[1].v, false, w2f[1][t].v,
                                                             (short)0, acc, false, false);
        }
        // residual (x = h + ff2 + b2) into xbuf [m][32]
        int idxm[8];
        #pragma unroll
        for (int r = 0; r < 8; ++r) idxm[r] = S.idx[r + 8 * hi];
        #pragma unroll
        for (int t = 0; t < 2; ++t) {
            int n = t * 16 + m16;
            float bb = b2f[n];
            #pragma unroll
            for (int r = 0; r < 8; ++r) {
                int m = r + 8 * hi;
                float hv = (float)embedH[idxm[r] * HID + n];
                S.xbuf[m * 32 + n] = c2[t].f[r] + bb + hv;
            }
        }
        // LayerNorm: lane m < 16 owns token m
        if (lane < 16) {
            const float* x = &S.xbuf[lane * 32];
            float s = 0.f, ss = 0.f;
            #pragma unroll
            for (int c = 0; c < 32; ++c) { float v = x[c]; s += v; ss += v * v; }
            float mu  = s * (1.f / 32.f);
            float var = ss * (1.f / 32.f) - mu * mu;
            float rs  = rsqrtf(var + 1e-5f);
            #pragma unroll
            for (int c = 0; c < 32; ++c)
                S.u.outb[lane * 32 + c] = (x[c] - mu) * rs * lngf[c] + lnbf[c];
        }
        // coalesced 2 KB store of 16 tokens x 32 floats
        #pragma unroll
        for (int q = 0; q < 4; ++q) {
            float4 v;
            __builtin_memcpy(&v, &S.u.outb[q * 128 + lane * 4], 16);
            __builtin_memcpy(&hidden[(size_t)tile * 512 + q * 128 + lane * 4], &v, 16);
        }
    }
}

// ---------------------------------------------------------------------------
// Kernel 2: energy-gated delta-rule scan. One wave32 per batch element; lane i
// holds row i of M (32 VGPRs); k broadcast via v_readlane into SGPRs.
// Gate needs only wave sums of {k^2, k*vp, vp^2}:
//   u.vp = a - p2/d,  |u|^2 = kk - 2a/d + p2/d^2,  d = kk + 1e-6
//   accept iff 2*u.vp + |u|^2*kk > 0   (== ||M+dM||^2 > ||M||^2)
// -> a single 3-value butterfly on the critical path per step.
// ---------------------------------------------------------------------------
__global__ __launch_bounds__(256) void scan_kernel(
    const float* __restrict__ hidden,
    const float* __restrict__ w_read, const float* __restrict__ b_read,
    const float* __restrict__ w_out,  const float* __restrict__ b_out,
    float* __restrict__ logits, float* __restrict__ counts)
{
    const int lane = threadIdx.x & 31;
    const int b = (blockIdx.x * blockDim.x + threadIdx.x) >> 5;
    if (b >= BATCH) return;
    const float* hb = hidden + (size_t)b * SEQLEN * HID;

    float M[32];
    #pragma unroll
    for (int j = 0; j < 32; ++j) M[j] = 0.f;

    float kv  = hb[lane];   // k at l = 0
    float cnt = 0.f;

    for (int l = 0; l < SEQLEN - 1; ++l) {
        float knext = hb[(l + 1) * HID + lane];           // prefetch next key

        float kb[32];                                     // uniform (SGPR) copies
        #pragma unroll
        for (int j = 0; j < 32; ++j) kb[j] = bcastf(kv, j);

        // vp_i = (M k)_i, 4 partial accumulators to shorten the FMA chain
        float vp0 = 0.f, vp1 = 0.f, vp2 = 0.f, vp3 = 0.f;
        #pragma unroll
        for (int j = 0; j < 32; j += 4) {
            vp0 = fmaf(M[j + 0], kb[j + 0], vp0);
            vp1 = fmaf(M[j + 1], kb[j + 1], vp1);
            vp2 = fmaf(M[j + 2], kb[j + 2], vp2);
            vp3 = fmaf(M[j + 3], kb[j + 3], vp3);
        }
        float vp = (vp0 + vp1) + (vp2 + vp3);

        // single butterfly over three per-lane products
        float kk = kv * kv, a = kv * vp, p2 = vp * vp;
        #pragma unroll
        for (int o = 16; o > 0; o >>= 1) {
            kk += __shfl_xor(kk, o, 32);
            a  += __shfl_xor(a,  o, 32);
            p2 += __shfl_xor(p2, o, 32);
        }
        float d  = kk + 1e-6f;
        float id = 1.f / d;
        float s1 = a - p2 * id;                       // u . vp
        float s2 = kk - 2.f * a * id + p2 * id * id;  // |u|^2
        float g  = (2.f * s1 + s2 * kk) > 0.f ? 1.f : 0.f;
        cnt += g;

        float u  = kv - vp * id;
        float gu = g * u;
        #pragma unroll
        for (int j = 0; j < 32; ++j) M[j] = fmaf(gu, kb[j], M[j]);  // rank-1 update
        kv = knext;
    }

    // kv == hidden[b, L-1, :]  -> ctx = M q
    float ctx = 0.f;
    #pragma unroll
    for (int j = 0; j < 32; ++j) ctx = fmaf(M[j], bcastf(kv, j), ctx);

    // r = ctx @ w_read + b_read  (lane owns output column)
    float r = b_read[lane];
    #pragma unroll
    for (int i = 0; i < 32; ++i) r = fmaf(bcastf(ctx, i), w_read[i * 32 + lane], r);

    // logits = r @ w_out + b_out (lane owns columns lane and lane+32)
    float o0 = b_out[lane], o1 = b_out[lane + 32];
    #pragma unroll
    for (int i = 0; i < 32; ++i) {
        float rb = bcastf(r, i);
        o0 = fmaf(rb, w_out[i * 64 + lane], o0);
        o1 = fmaf(rb, w_out[i * 64 + 32 + lane], o1);
    }
    logits[b * 64 + lane]      = o0;
    logits[b * 64 + 32 + lane] = o1;
    if (lane == 0) counts[b] = cnt;
}

// ---------------------------------------------------------------------------
// Kernel 3: reduce gate counts -> write_rate at d_out[B*V]
// ---------------------------------------------------------------------------
__global__ void reduce_kernel(const float* __restrict__ counts, float* __restrict__ out)
{
    const int lane = threadIdx.x;
    float s = 0.f;
    for (int i = lane; i < BATCH; i += 32) s += counts[i];
    #pragma unroll
    for (int o = 16; o > 0; o >>= 1) s += __shfl_xor(s, o, 32);
    if (lane == 0) out[BATCH * VOC] = s / (float)((SEQLEN - 1) * BATCH);
}

extern "C" void kernel_launch(void* const* d_in, const int* in_sizes, int n_in,
                              void* d_out, int out_size, void* d_ws, size_t ws_size,
                              hipStream_t stream) {
    (void)in_sizes; (void)n_in; (void)out_size; (void)ws_size;
    const int*   seq    = (const int*)  d_in[0];
    const float* embed  = (const float*)d_in[1];
    const float* w1     = (const float*)d_in[2];
    const float* b1     = (const float*)d_in[3];
    const float* w2     = (const float*)d_in[4];
    const float* b2     = (const float*)d_in[5];
    const float* ln_g   = (const float*)d_in[6];
    const float* ln_b   = (const float*)d_in[7];
    const float* w_read = (const float*)d_in[8];
    const float* b_read = (const float*)d_in[9];
    const float* w_out  = (const float*)d_in[10];
    const float* b_out  = (const float*)d_in[11];

    float* out    = (float*)d_out;
    float* counts = (float*)d_ws;                           // 512 floats
    float* hidden = (float*)((char*)d_ws + 4096);           // B*L*H f32 = 128 MB

    encoder_kernel<<<1024, 256, 0, stream>>>(seq, embed, w1, b1, w2, b2, ln_g, ln_b, hidden);
    scan_kernel<<<(BATCH * 32 + 255) / 256, 256, 0, stream>>>(hidden, w_read, b_read, w_out, b_out, out, counts);
    reduce_kernel<<<1, 32, 0, stream>>>(counts, out);
}
